// Model_22368189677787
// MI455X (gfx1250) — compile-verified, matrix operand order checked by reading, not run on hardware
//
#include <hip/hip_runtime.h>
#include <hip/hip_bf16.h>
#include <math.h>

// ---------------------------------------------------------------------------
// Problem constants (match reference)
// ---------------------------------------------------------------------------
#define N_NODES   100000
#define N_EDGES   1600000
#define F_IN      64
#define HID       128
#define N_GRAPHS  256
#define MPNN_STEPS 4
#define S2S_STEPS  3

typedef __attribute__((ext_vector_type(16))) __bf16 v16bf;
typedef __attribute__((ext_vector_type(8)))  __bf16 v8bf;
typedef __attribute__((ext_vector_type(8)))  float  v8f;

// ---------------------------------------------------------------------------
// fp32 -> bf16 conversion (one-time, 25.6MB pass)
// ---------------------------------------------------------------------------
__global__ void k_to_bf16(const float* __restrict__ in, __bf16* __restrict__ out, int n) {
    int i = blockIdx.x * blockDim.x + threadIdx.x;
    if (i < n) out[i] = (__bf16)in[i];
}

// ---------------------------------------------------------------------------
// Input layer: h = relu(x @ W_in^T + b_in) via v_wmma_f32_16x16x32_bf16
//   x: [N,64] bf16 (row-major), W_in: [128,64] bf16 (row-major; acts as B^T
//   so B[k][n] = W_in[n][k] -> contiguous in k per output column)
//   Grid: 6250 M-tiles; block 256 = 8 waves; wave w handles N-tile w (16 cols).
//   Fragment layout per CDNA5 ISA 7.12.2 (16-bit A 16x32; B symmetric; C/D v8f).
// ---------------------------------------------------------------------------
__global__ void k_input_wmma(const __bf16* __restrict__ xb,
                             const __bf16* __restrict__ wb,
                             const float*  __restrict__ b_in,
                             float* __restrict__ h) {
    const int mtile = blockIdx.x;            // 0..6249
    const int wave  = threadIdx.x >> 5;      // 0..7  -> N tile
    const int lane  = threadIdx.x & 31;
    const int l16   = lane & 15;
    const int hi    = lane >> 4;             // 0 or 1 (half-wave)
    const int row   = mtile * 16 + l16;      // M index owned by this lane (A)
    const int col   = wave  * 16 + l16;      // N index owned by this lane (B/C/D)
    const int kbase = hi * 8;                // K sub-block per half-wave

    v8f acc = {};
#pragma unroll
    for (int kt = 0; kt < 2; ++kt) {         // K = 64 = 2 x 32
        const int k0 = kt * 32 + kbase;
        // A: lane holds A[row][k0..k0+7] and A[row][k0+16..k0+23] (16B each, aligned)
        const v8bf alo = *(const v8bf*)(xb + (size_t)row * F_IN + k0);
        const v8bf ahi = *(const v8bf*)(xb + (size_t)row * F_IN + k0 + 16);
        // B: lane holds B[k0..k0+7][col], B[k0+16..k0+23][col] == W_in[col][k...]
        const v8bf blo = *(const v8bf*)(wb + (size_t)col * F_IN + k0);
        const v8bf bhi = *(const v8bf*)(wb + (size_t)col * F_IN + k0 + 16);
        const v16bf a = __builtin_shufflevector(alo, ahi, 0,1,2,3,4,5,6,7,8,9,10,11,12,13,14,15);
        const v16bf b = __builtin_shufflevector(blo, bhi, 0,1,2,3,4,5,6,7,8,9,10,11,12,13,14,15);
        acc = __builtin_amdgcn_wmma_f32_16x16x32_bf16(
            /*neg_a=*/false, a, /*neg_b=*/false, b,
            /*c_mod=*/(short)0, acc, /*reuse_a=*/false, /*reuse_b=*/false);
    }
    const float bias = b_in[col];
#pragma unroll
    for (int r = 0; r < 8; ++r) {            // C/D: VGPR r -> M = r (+8 hi half)
        const int mrow = mtile * 16 + r + hi * 8;
        h[(size_t)mrow * HID + col] = fmaxf(acc[r] + bias, 0.0f);
    }
}

// ---------------------------------------------------------------------------
// MPNN
// ---------------------------------------------------------------------------
__global__ void k_zero_m_deg(float* __restrict__ m, float* __restrict__ deg) {
    int i = blockIdx.x * blockDim.x + threadIdx.x;
    if (i < N_NODES * HID) m[i] = 0.0f;
    if (i < N_NODES) deg[i] = 0.0f;
}

__global__ void k_degree(const int* __restrict__ dst, float* __restrict__ deg) {
    int e = blockIdx.x * blockDim.x + threadIdx.x;
    if (e < N_EDGES) atomicAdd(deg + dst[e], 1.0f);
}

// wave-per-edge gather/scatter: lane owns a float4 chunk of the 128-feature row.
// h (51MB) and m (51MB) are L2-resident on MI455X (192MB L2) -> L2-bound.
__global__ void k_scatter(const int* __restrict__ src, const int* __restrict__ dst,
                          const float* __restrict__ h, float* __restrict__ m) {
    const int edge = (blockIdx.x * blockDim.x + threadIdx.x) >> 5;
    const int lane = threadIdx.x & 31;
    if (edge >= N_EDGES) return;
    const int s = src[edge];
    const int d = dst[edge];
    const float4 hv = *(const float4*)(h + (size_t)s * HID + lane * 4);
    float* mp = m + (size_t)d * HID + lane * 4;
    atomicAdd(mp + 0, hv.x);
    atomicAdd(mp + 1, hv.y);
    atomicAdd(mp + 2, hv.z);
    atomicAdd(mp + 3, hv.w);
}

// h = (h + m/deg) * 0.5 ; m = 0 (ready for next step)
__global__ void k_update(float* __restrict__ h, float* __restrict__ m,
                         const float* __restrict__ deg) {
    int i = blockIdx.x * blockDim.x + threadIdx.x;
    if (i >= N_NODES * HID) return;
    const float d = fmaxf(deg[i >> 7], 1.0f);
    h[i] = (h[i] + m[i] / d) * 0.5f;
    m[i] = 0.0f;
}

// ---------------------------------------------------------------------------
// Set2Set
// ---------------------------------------------------------------------------
__global__ void k_s2s_init(float* __restrict__ q_star, float* __restrict__ hl,
                           float* __restrict__ cl) {
    int i = blockIdx.x * blockDim.x + threadIdx.x;
    if (i < N_GRAPHS * 2 * HID) q_star[i] = 0.0f;
    if (i < N_GRAPHS * HID) { hl[i] = 0.0f; cl[i] = 0.0f; }
}

// gates[g][j] = q_star[g]·W_ih[j] + hl[g]·W_hh[j] + b_ih[j] + b_hh[j]
__global__ void k_lstm_gates(const float* __restrict__ q_star, const float* __restrict__ hl,
                             const float* __restrict__ W_ih, const float* __restrict__ W_hh,
                             const float* __restrict__ b_ih, const float* __restrict__ b_hh,
                             float* __restrict__ gates) {
    int idx = blockIdx.x * blockDim.x + threadIdx.x;
    if (idx >= N_GRAPHS * 4 * HID) return;
    const int g = idx >> 9;          // graph
    const int j = idx & 511;         // gate row
    const float* qs = q_star + (size_t)g * 2 * HID;
    const float* wi = W_ih + (size_t)j * 2 * HID;
    float acc = b_ih[j] + b_hh[j];
#pragma unroll 4
    for (int k = 0; k < 2 * HID; ++k) acc += qs[k] * wi[k];
    const float* hv = hl   + (size_t)g * HID;
    const float* wh = W_hh + (size_t)j * HID;
#pragma unroll 4
    for (int k = 0; k < HID; ++k) acc += hv[k] * wh[k];
    gates[idx] = acc;
}

__device__ __forceinline__ float sigmoidf(float x) { return 1.0f / (1.0f + expf(-x)); }

__global__ void k_lstm_point(const float* __restrict__ gates,
                             float* __restrict__ cl, float* __restrict__ hl) {
    int idx = blockIdx.x * blockDim.x + threadIdx.x;
    if (idx >= N_GRAPHS * HID) return;
    const int g = idx >> 7, j = idx & (HID - 1);
    const float* gg = gates + (size_t)g * 4 * HID;
    const float iv = sigmoidf(gg[j]);
    const float fv = sigmoidf(gg[HID + j]);
    const float gv = tanhf(gg[2 * HID + j]);
    const float ov = sigmoidf(gg[3 * HID + j]);
    const float c  = fv * cl[idx] + iv * gv;
    cl[idx] = c;
    hl[idx] = ov * tanhf(c);
}

__global__ void k_attn_init(float* __restrict__ r_un, float* __restrict__ asum,
                            float* __restrict__ emax) {
    int i = blockIdx.x * blockDim.x + threadIdx.x;
    if (i < N_GRAPHS * HID) r_un[i] = 0.0f;
    if (i < N_GRAPHS) { asum[i] = 0.0f; emax[i] = -3.402823466e38f; }
}

// sign-aware int-punned float atomic max (works for all finite floats)
__device__ __forceinline__ void atomicMaxF(float* addr, float val) {
    if (val >= 0.0f) atomicMax((int*)addr, __float_as_int(val));
    else             atomicMin((unsigned int*)addr, (unsigned int)__float_as_int(val));
}

// wave-per-node: e[n] = h[n]·q[batch[n]]  + segment max
__global__ void k_attn_e(const float* __restrict__ h, const float* __restrict__ hl,
                         const int* __restrict__ batch,
                         float* __restrict__ e, float* __restrict__ emax) {
    const int node = (blockIdx.x * blockDim.x + threadIdx.x) >> 5;
    const int lane = threadIdx.x & 31;
    if (node >= N_NODES) return;
    const int g = batch[node];
    const float4 hv = *(const float4*)(h  + (size_t)node * HID + lane * 4);
    const float4 qv = *(const float4*)(hl + (size_t)g    * HID + lane * 4);
    float p = hv.x * qv.x + hv.y * qv.y + hv.z * qv.z + hv.w * qv.w;
#pragma unroll
    for (int off = 16; off; off >>= 1) p += __shfl_xor(p, off, 32);
    if (lane == 0) { e[node] = p; atomicMaxF(emax + g, p); }
}

// wave-per-node: a = exp(e - emax); asum += a; r_un += a*h[n]
__global__ void k_attn_acc(const float* __restrict__ h, const float* __restrict__ e,
                           const float* __restrict__ emax, const int* __restrict__ batch,
                           float* __restrict__ asum, float* __restrict__ r_un) {
    const int node = (blockIdx.x * blockDim.x + threadIdx.x) >> 5;
    const int lane = threadIdx.x & 31;
    if (node >= N_NODES) return;
    const int g = batch[node];
    const float a = expf(e[node] - emax[g]);
    if (lane == 0) atomicAdd(asum + g, a);
    const float4 hv = *(const float4*)(h + (size_t)node * HID + lane * 4);
    float* rp = r_un + (size_t)g * HID + lane * 4;
    atomicAdd(rp + 0, a * hv.x);
    atomicAdd(rp + 1, a * hv.y);
    atomicAdd(rp + 2, a * hv.z);
    atomicAdd(rp + 3, a * hv.w);
}

// q_star = [hl , r_un/asum]
__global__ void k_attn_fin(const float* __restrict__ hl, const float* __restrict__ r_un,
                           const float* __restrict__ asum, float* __restrict__ q_star) {
    int idx = blockIdx.x * blockDim.x + threadIdx.x;
    if (idx >= N_GRAPHS * HID) return;
    const int g = idx >> 7, j = idx & (HID - 1);
    q_star[(size_t)g * 2 * HID + j]       = hl[idx];
    q_star[(size_t)g * 2 * HID + HID + j] = r_un[idx] / asum[g];
}

// out[g] = q_star[g]·W_pred + b_pred  (wave per graph)
__global__ void k_predict(const float* __restrict__ q_star, const float* __restrict__ W_pred,
                          const float* __restrict__ b_pred, float* __restrict__ out) {
    const int g = (blockIdx.x * blockDim.x + threadIdx.x) >> 5;
    const int lane = threadIdx.x & 31;
    if (g >= N_GRAPHS) return;
    const float* qs = q_star + (size_t)g * 2 * HID;
    float p = 0.0f;
#pragma unroll
    for (int k = lane; k < 2 * HID; k += 32) p += qs[k] * W_pred[k];
#pragma unroll
    for (int off = 16; off; off >>= 1) p += __shfl_xor(p, off, 32);
    if (lane == 0) out[g] = p + b_pred[0];
}

// ---------------------------------------------------------------------------
// Host launcher
// ---------------------------------------------------------------------------
extern "C" void kernel_launch(void* const* d_in, const int* in_sizes, int n_in,
                              void* d_out, int out_size, void* d_ws, size_t ws_size,
                              hipStream_t stream) {
    (void)in_sizes; (void)n_in; (void)out_size; (void)ws_size;

    const float* x      = (const float*)d_in[0];
    const int*   ei     = (const int*)  d_in[1];   // [2, E] : row0 = src, row1 = dst
    const int*   batch  = (const int*)  d_in[2];
    const float* W_in   = (const float*)d_in[3];
    const float* b_in   = (const float*)d_in[4];
    const float* W_ih   = (const float*)d_in[5];
    const float* W_hh   = (const float*)d_in[6];
    const float* b_ih   = (const float*)d_in[7];
    const float* b_hh   = (const float*)d_in[8];
    const float* W_pred = (const float*)d_in[9];
    const float* b_pred = (const float*)d_in[10];
    float* out = (float*)d_out;

    const int* src = ei;
    const int* dst = ei + N_EDGES;

    // --- workspace carve-up (~117 MB, all 16B-aligned) ---
    char* p = (char*)d_ws;
    float*  h      = (float*)p;  p += (size_t)N_NODES * HID * sizeof(float);
    float*  m      = (float*)p;  p += (size_t)N_NODES * HID * sizeof(float);
    float*  deg    = (float*)p;  p += (size_t)N_NODES * sizeof(float);
    float*  ev     = (float*)p;  p += (size_t)N_NODES * sizeof(float);
    __bf16* xb     = (__bf16*)p; p += (size_t)N_NODES * F_IN * sizeof(__bf16);
    __bf16* wb     = (__bf16*)p; p += (size_t)HID * F_IN * sizeof(__bf16);
    float*  gates  = (float*)p;  p += (size_t)N_GRAPHS * 4 * HID * sizeof(float);
    float*  hl     = (float*)p;  p += (size_t)N_GRAPHS * HID * sizeof(float);
    float*  cl     = (float*)p;  p += (size_t)N_GRAPHS * HID * sizeof(float);
    float*  q_star = (float*)p;  p += (size_t)N_GRAPHS * 2 * HID * sizeof(float);
    float*  r_un   = (float*)p;  p += (size_t)N_GRAPHS * HID * sizeof(float);
    float*  asum   = (float*)p;  p += (size_t)N_GRAPHS * sizeof(float);
    float*  emax   = (float*)p;  p += (size_t)N_GRAPHS * sizeof(float);

    const int B = 256;

    // 1) bf16 conversion of GEMM operands
    k_to_bf16<<<(N_NODES * F_IN + B - 1) / B, B, 0, stream>>>(x, xb, N_NODES * F_IN);
    k_to_bf16<<<(HID * F_IN + B - 1) / B, B, 0, stream>>>(W_in, wb, HID * F_IN);

    // 2) input layer GEMM on the matrix pipe (WMMA bf16 -> f32)
    k_input_wmma<<<N_NODES / 16, B, 0, stream>>>(xb, wb, b_in, h);

    // 3) degrees + zero message buffer
    k_zero_m_deg<<<(N_NODES * HID + B - 1) / B, B, 0, stream>>>(m, deg);
    k_degree<<<(N_EDGES + B - 1) / B, B, 0, stream>>>(dst, deg);

    // 4) MPNN steps (L2-resident gather/scatter)
    for (int s = 0; s < MPNN_STEPS; ++s) {
        k_scatter<<<(N_EDGES * 32 + B - 1) / B, B, 0, stream>>>(src, dst, h, m);
        k_update<<<(N_NODES * HID + B - 1) / B, B, 0, stream>>>(h, m, deg);
    }

    // 5) Set2Set readout
    k_s2s_init<<<(N_GRAPHS * 2 * HID + B - 1) / B, B, 0, stream>>>(q_star, hl, cl);
    for (int s = 0; s < S2S_STEPS; ++s) {
        k_lstm_gates<<<(N_GRAPHS * 4 * HID + B - 1) / B, B, 0, stream>>>(
            q_star, hl, W_ih, W_hh, b_ih, b_hh, gates);
        k_lstm_point<<<(N_GRAPHS * HID + B - 1) / B, B, 0, stream>>>(gates, cl, hl);
        k_attn_init<<<(N_GRAPHS * HID + B - 1) / B, B, 0, stream>>>(r_un, asum, emax);
        k_attn_e<<<(N_NODES * 32 + B - 1) / B, B, 0, stream>>>(h, hl, batch, ev, emax);
        k_attn_acc<<<(N_NODES * 32 + B - 1) / B, B, 0, stream>>>(h, ev, emax, batch, asum, r_un);
        k_attn_fin<<<(N_GRAPHS * HID + B - 1) / B, B, 0, stream>>>(hl, r_un, asum, q_star);
    }

    // 6) prediction head
    k_predict<<<(N_GRAPHS * 32 + B - 1) / B, B, 0, stream>>>(q_star, W_pred, b_pred, out);
}
